// TimeSeriesGCN_79499844649193
// MI455X (gfx1250) — matrix operand order repaired and checked by reading, not compiled, not run on hardware
//
#include <hip/hip_runtime.h>
#include <hip/hip_bf16.h>

typedef _Float16 half_t;
typedef __attribute__((ext_vector_type(16))) _Float16 v16h;
typedef __attribute__((ext_vector_type(8)))  float    v8f;
typedef __attribute__((ext_vector_type(4)))  float    v4f;
typedef __attribute__((ext_vector_type(4)))  int      i4v;

// ---------------- problem constants ----------------
#define NB    8192
#define NNODE 96
#define CIN0  16

// ---------------- optional CDNA5 async global->LDS path (compile-safe guards) ----
#if defined(__has_builtin)
#if __has_builtin(__builtin_amdgcn_global_load_async_to_lds_b128) && \
    __has_builtin(__builtin_amdgcn_s_wait_asynccnt)
#define USE_ASYNC_LDS 1
#endif
#endif

typedef __attribute__((address_space(1))) i4v as1_i4;
typedef __attribute__((address_space(3))) i4v as3_i4;

__device__ __forceinline__ void g2l_b128(const half_t* g, half_t* l) {
#ifdef USE_ASYNC_LDS
    __builtin_amdgcn_global_load_async_to_lds_b128(
        (as1_i4*)(unsigned long long)g,
        (as3_i4*)(unsigned)(unsigned long long)l, 0, 0);
#else
    *(v4f*)l = *(const v4f*)g;
#endif
}
__device__ __forceinline__ void async_wait() {
#ifdef USE_ASYNC_LDS
    __builtin_amdgcn_s_wait_asynccnt(0);
#endif
}

// ---------------- workspace layout (in halves) ----------------
constexpr int OFF_AHAT = 0;                       // 96*96
constexpr int OFF_W1T  = OFF_AHAT + 96 * 96;      // [64][32]
constexpr int OFF_W2T  = OFF_W1T + 64 * 32;       // [32][64]
constexpr int OFF_C1W  = OFF_W2T + 32 * 64;       // [64][224]
constexpr int OFF_C2W  = OFF_C1W + 64 * 224;      // [128][320]
constexpr int OFF_C3W  = OFF_C2W + 128 * 320;     // [256][384]
constexpr int OFF_FC1T = OFF_C3W + 256 * 384;     // [128][3072]
constexpr int OFF_CL1T = OFF_FC1T + 128 * 3072;   // [128][128]
constexpr int OFF_CL2T = OFF_CL1T + 128 * 128;    // [224][128] (210 padded)

// ---------------- dynamic LDS layout (in halves), 8 waves ----------------
constexpr int SM_AHAT   = 0;                      // 9216
constexpr int SM_U      = 9216;                   // 8 waves * 12288
constexpr int SM_A4     = SM_U + 8 * 12288;       // 16 * 3072
constexpr int SM_R1     = SM_A4 + 16 * 3072;      // 16 * 128
constexpr int SM_R2     = SM_R1 + 16 * 128;       // 16 * 128
constexpr int SM_HALVES = SM_R2 + 16 * 128;       // 160768 halves
constexpr size_t SMEM_BYTES = (size_t)SM_HALVES * 2;  // 321536 B (< 320 KB WG limit)

// per-wave scratch (halves within U, 12288 = 24KB)
constexpr int U_T1  = 0;      // [96][32]
constexpr int U_H1  = 3072;   // [96][64]
constexpr int U_T2T = 0;      // [32][96] (aliases T1, dead)
constexpr int U_H2  = 9216;   // [32][96] conv1 input
constexpr int U_IC1 = 0;      // [16][224]
constexpr int U_C1O = 3584;   // [64][48]
constexpr int U_IC2 = 6656;   // [16][320]
constexpr int U_C2O = 0;      // [128][24]
constexpr int U_IC3 = 3584;   // [16][384]
constexpr int U_SIZE = 12288;

// ---------------- WMMA helpers ----------------
union V16HU { v16h v; v4f f[2]; };

__device__ __forceinline__ v8f vzero8() {
    v8f z = {0.f, 0.f, 0.f, 0.f, 0.f, 0.f, 0.f, 0.f};
    return z;
}

__device__ __forceinline__ v8f wmma16(v16h a, v16h b, v8f c) {
    return __builtin_amdgcn_wmma_f32_16x16x32_f16(false, a, false, b, (short)0, c, false, false);
}

// A operand: row-major [M][K]; lane l -> row l&15; 16 contiguous K halves = hi*16+{0..15}
__device__ __forceinline__ v16h load_tileA(const half_t* base, int strideH, int lane) {
    int m = lane & 15, hi = (lane >> 4) & 1;
    const half_t* p = base + m * strideH + hi * 16;
    V16HU u;
    u.f[0] = *(const v4f*)(p);
    u.f[1] = *(const v4f*)(p + 8);
    return u.v;
}

// B operand: stored [N][K] row-major; lane l -> col n=l&15; 16 contiguous K halves
__device__ __forceinline__ v16h load_tileB(const half_t* base, int strideH, int lane) {
    int n = lane & 15, kb = (lane >> 4) * 16;
    const half_t* p = base + n * strideH + kb;
    V16HU u;
    u.f[0] = *(const v4f*)(p);
    u.f[1] = *(const v4f*)(p + 8);
    return u.v;
}

// B operand gathered from fp32 row-major [N][K]
__device__ __forceinline__ v16h load_tileB_f32(const float* base, int strideF, int lane) {
    int n = lane & 15, kb = (lane >> 4) * 16;
    const float* p = base + n * strideF + kb;
    v16h r;
#pragma unroll
    for (int j = 0; j < 16; ++j) r[j] = (half_t)p[j];
    return r;
}

__device__ __forceinline__ unsigned pack2h(float v0, float v1, bool relu) {
    if (relu) { v0 = fmaxf(v0, 0.f); v1 = fmaxf(v1, 0.f); }
    union { half_t h[2]; unsigned u; } cv;
    cv.h[0] = (half_t)v0; cv.h[1] = (half_t)v1;
    return cv.u;
}

// D row-major store (scalar): element (m,n) -> base[m*strideH + n]
__device__ __forceinline__ void store_tile_f16(half_t* base, int strideH, v8f acc,
                                               const float* bias, int n0, bool relu, int lane) {
    int n = lane & 15, mh = (lane >> 4) * 8;
    float bv = bias ? bias[n0 + n] : 0.0f;
#pragma unroll
    for (int r = 0; r < 8; ++r) {
        float v = acc[r] + bv;
        if (relu) v = fmaxf(v, 0.0f);
        base[(mh + r) * strideH + n] = (half_t)v;
    }
}

// transposed store, packed pairs, bias indexed by column n of D
__device__ __forceinline__ void store_tile_T_bn(half_t* base, int strideH, v8f acc,
                                                const float* bias, int n0, bool relu,
                                                int validRows, int lane) {
    int n = lane & 15, mh = (lane >> 4) * 8;
    float bv = bias ? bias[n0 + n] : 0.0f;
    unsigned w[4];
#pragma unroll
    for (int p = 0; p < 4; ++p) w[p] = pack2h(acc[2 * p] + bv, acc[2 * p + 1] + bv, relu);
    unsigned* p32 = (unsigned*)(base + n * strideH + mh);
    int npairs = (validRows - mh) >> 1;
    if (npairs > 4) npairs = 4;
#pragma unroll
    for (int p = 0; p < 4; ++p)
        if (p < npairs) p32[p] = w[p];
}

// transposed store, packed pairs, bias indexed by row m of D
__device__ __forceinline__ void store_tile_T_bm(half_t* base, int strideH, v8f acc,
                                                const float* bias, int m0, bool relu, int lane) {
    int n = lane & 15, mh = (lane >> 4) * 8;
    unsigned w[4];
#pragma unroll
    for (int p = 0; p < 4; ++p) {
        float b0 = bias[m0 + mh + 2 * p];
        float b1 = bias[m0 + mh + 2 * p + 1];
        w[p] = pack2h(acc[2 * p] + b0, acc[2 * p + 1] + b1, relu);
    }
    unsigned* p32 = (unsigned*)(base + n * strideH + mh);
#pragma unroll
    for (int p = 0; p < 4; ++p) p32[p] = w[p];
}

__device__ __forceinline__ void store_tile_f32(float* base, int strideF, v8f acc,
                                               const float* bias, int n0, bool relu,
                                               int validCols, int lane) {
    int n = lane & 15, mh = (lane >> 4) * 8;
    if (n >= validCols) return;
    float bv = bias[n0 + n];
#pragma unroll
    for (int r = 0; r < 8; ++r) {
        float v = acc[r] + bv;
        if (relu) v = fmaxf(v, 0.0f);
        base[(size_t)(mh + r) * strideF + n] = v;
    }
}

// ---------------- prep kernel 1: dense normalized adjacency (f16) ----------------
__global__ void k_prep_adj(const int* __restrict__ edge, half_t* __restrict__ AhatH) {
    __shared__ float deg[96];
    __shared__ float dinv[96];
    __shared__ float Af[96 * 96];
    int t = threadIdx.x, nt = blockDim.x;
    for (int i = t; i < 96; i += nt) deg[i] = 1.0f;
    for (int i = t; i < 96 * 96; i += nt) Af[i] = 0.0f;
    __syncthreads();
    for (int e = t; e < 768; e += nt) atomicAdd(&deg[edge[e]], 1.0f);
    __syncthreads();
    for (int i = t; i < 96; i += nt) dinv[i] = rsqrtf(fmaxf(deg[i], 1e-12f));
    __syncthreads();
    for (int e = t; e < 768; e += nt) {
        int r = edge[e], c = edge[768 + e];
        atomicAdd(&Af[r * 96 + c], dinv[r] * dinv[c]);
    }
    for (int i = t; i < 96; i += nt) atomicAdd(&Af[i * 96 + i], dinv[i] * dinv[i]);
    __syncthreads();
    for (int i = t; i < 96 * 96; i += nt) AhatH[i] = (half_t)Af[i];
}

// ---------------- prep kernel 2: weights -> f16 [N][K] layouts ----------------
__global__ void k_prep_weights(const float* __restrict__ g1w, const float* __restrict__ g2w,
                               const float* __restrict__ c1w, const float* __restrict__ c2w,
                               const float* __restrict__ c3w, const float* __restrict__ f1w,
                               const float* __restrict__ l1w, const float* __restrict__ l2w,
                               half_t* __restrict__ wsh) {
    int tid = blockIdx.x * blockDim.x + threadIdx.x;
    int np = gridDim.x * blockDim.x;
    half_t* W1t = wsh + OFF_W1T;
    half_t* W2t = wsh + OFF_W2T;
    half_t* C1 = wsh + OFF_C1W;
    half_t* C2 = wsh + OFF_C2W;
    half_t* C3 = wsh + OFF_C3W;
    half_t* F1 = wsh + OFF_FC1T;
    half_t* L1 = wsh + OFF_CL1T;
    half_t* L2 = wsh + OFF_CL2T;
    for (int i = tid; i < 64 * 32; i += np) {
        int n = i / 32, k = i - n * 32;
        W1t[i] = (half_t)((k < 16) ? g1w[k * 64 + n] : 0.0f);
    }
    for (int i = tid; i < 32 * 64; i += np) {
        int n = i / 64, k = i - n * 64;
        W2t[i] = (half_t)g2w[k * 32 + n];
    }
    for (int i = tid; i < 64 * 224; i += np)  C1[i] = (half_t)c1w[i];
    for (int i = tid; i < 128 * 320; i += np) C2[i] = (half_t)c2w[i];
    for (int i = tid; i < 256 * 384; i += np) C3[i] = (half_t)c3w[i];
    for (int i = tid; i < 128 * 3072; i += np) {
        int n = i / 3072, k = i - n * 3072;
        F1[i] = (half_t)f1w[k * 128 + n];
    }
    for (int i = tid; i < 128 * 128; i += np) {
        int n = i >> 7, k = i & 127;
        L1[i] = (half_t)l1w[k * 128 + n];
    }
    for (int i = tid; i < 224 * 128; i += np) {
        int n = i >> 7, k = i & 127;
        L2[i] = (half_t)((n < 210) ? l2w[k * 210 + n] : 0.0f);
    }
}

// ---------------- implicit-GEMM conv1d (per-wave, operands/output in LDS) ----------------
template <int CI, int LI, int CO, int LO, int KW, int ST, int PD>
__device__ void conv_layer(const half_t* __restrict__ xin,  // LDS [CI][LI]
                           half_t* __restrict__ imcol,      // LDS [16][KK]
                           half_t* __restrict__ outT,       // LDS [CO][LO]
                           const half_t* __restrict__ wt,   // global [CO][KK]
                           const float* __restrict__ bias, int lane) {
    constexpr int KK = CI * KW;          // multiple of 32
    constexpr int NT = CO / 16;
    constexpr int CHUNK = (NT < 8) ? NT : 8;
    for (int lt0 = 0; lt0 < LO; lt0 += 16) {
        for (int idx = lane; idx < 16 * KK; idx += 32) {
            int rr = idx / KK, kk = idx - rr * KK;
            int l = lt0 + rr;
            int c = kk / KW, kw = kk - c * KW;
            int li = l * ST - PD + kw;
            half_t v = (half_t)0;
            if (l < LO && li >= 0 && li < LI) v = xin[c * LI + li];
            imcol[rr * KK + kk] = v;
        }
        int validRows = (LO - lt0 < 16) ? (LO - lt0) : 16;
        for (int nc = 0; nc < NT; nc += CHUNK) {
            v8f acc[CHUNK];
#pragma unroll
            for (int j = 0; j < CHUNK; ++j) acc[j] = vzero8();
            for (int kt = 0; kt < KK / 32; ++kt) {
                v16h a = load_tileA(imcol + kt * 32, KK, lane);
#pragma unroll
                for (int j = 0; j < CHUNK; ++j) {
                    v16h bm = load_tileB(wt + (nc + j) * 16 * KK + kt * 32, KK, lane);
                    acc[j] = wmma16(a, bm, acc[j]);
                }
            }
#pragma unroll
            for (int j = 0; j < CHUNK; ++j)
                store_tile_T_bn(outT + (nc + j) * 16 * LO + lt0, LO, acc[j],
                                bias, (nc + j) * 16, true, validRows, lane);
        }
    }
}

// ---------------- fused main kernel ----------------
__global__ __launch_bounds__(256, 1)
void k_fused(const float* __restrict__ src,
             const float* __restrict__ gcn1_b, const float* __restrict__ gcn2_b,
             const float* __restrict__ conv1_b, const float* __restrict__ conv2_b,
             const float* __restrict__ conv3_b, const float* __restrict__ fc1_b,
             const float* __restrict__ cls1_b, const float* __restrict__ cls2_b,
             const half_t* __restrict__ wsh,
             float* __restrict__ outCls, float* __restrict__ outFeat) {
    extern __shared__ char smem_raw[];
    half_t* sAhat = (half_t*)smem_raw;
    half_t* sU = sAhat + SM_U;
    half_t* sA4 = sAhat + SM_A4;
    half_t* sR1 = sAhat + SM_R1;
    half_t* sR2 = sAhat + SM_R2;

    const half_t* wsW1t = wsh + OFF_W1T;
    const half_t* wsW2t = wsh + OFF_W2T;
    const half_t* wsC1 = wsh + OFF_C1W;
    const half_t* wsC2 = wsh + OFF_C2W;
    const half_t* wsC3 = wsh + OFF_C3W;
    const half_t* wsF1 = wsh + OFF_FC1T;
    const half_t* wsL1 = wsh + OFF_CL1T;
    const half_t* wsL2 = wsh + OFF_CL2T;

    int lane = threadIdx.x & 31;
    int wv = threadIdx.x >> 5;

    // stage Ahat into LDS (async global->LDS on this toolchain)
    for (int t = threadIdx.x; t < 96 * 96 / 8; t += blockDim.x)
        g2l_b128(wsh + OFF_AHAT + t * 8, sAhat + t * 8);
    async_wait();
    __syncthreads();

    half_t* U = sU + wv * U_SIZE;
    int b0g = blockIdx.x * 16;

    // ---- phase 1: per-wave, 2 batches each through GCN + convs ----
    for (int i = 0; i < 2; ++i) {
        int arow = wv * 2 + i;
        int b = b0g + arow;
        const float* srcb = src + (size_t)b * (CIN0 * NNODE);
        half_t* T1 = U + U_T1;
        half_t* H1 = U + U_H1;
        half_t* T2t = U + U_T2T;
        half_t* H2 = U + U_H2;

        // GEMM1: T1 = Â @ X   (M=96, K=96, N=16); B tiles hoisted, converted once
        {
            v16h xb[3];
#pragma unroll
            for (int kt = 0; kt < 3; ++kt) xb[kt] = load_tileB_f32(srcb + kt * 32, 96, lane);
            for (int mt = 0; mt < 6; ++mt) {
                v8f acc = vzero8();
#pragma unroll
                for (int kt = 0; kt < 3; ++kt) {
                    v16h a = load_tileA(sAhat + mt * 16 * 96 + kt * 32, 96, lane);
                    acc = wmma16(a, xb[kt], acc);
                }
                store_tile_f16(T1 + mt * 16 * 32, 32, acc, nullptr, 0, false, lane);
            }
            for (int t = lane; t < 96 * 16; t += 32)
                T1[(t >> 4) * 32 + 16 + (t & 15)] = (half_t)0;  // zero K pad
        }

        // GEMM2 (flipped): H1^T = W1^T @ T1^T -> H1 stored row-major [node][64]
        {
            v16h wa[4];
#pragma unroll
            for (int mt = 0; mt < 4; ++mt) wa[mt] = load_tileA(wsW1t + mt * 16 * 32, 32, lane);
            for (int ntile = 0; ntile < 6; ++ntile) {
                v16h bm = load_tileB(T1 + ntile * 16 * 32, 32, lane);
#pragma unroll
                for (int mt = 0; mt < 4; ++mt) {
                    v8f acc = wmma16(wa[mt], bm, vzero8());
                    store_tile_T_bm(H1 + ntile * 16 * 64 + mt * 16, 64, acc,
                                    gcn1_b, mt * 16, true, lane);
                }
            }
        }

        // GEMM3: T2t = (H1 @ W2)^T   (A = H1 row-major, B = W2t; contiguous T store)
        {
            v16h wb[4];
#pragma unroll
            for (int ntile = 0; ntile < 2; ++ntile)
#pragma unroll
                for (int kt = 0; kt < 2; ++kt)
                    wb[ntile * 2 + kt] = load_tileB(wsW2t + ntile * 16 * 64 + kt * 32, 64, lane);
            for (int mt = 0; mt < 6; ++mt) {
                v16h a0 = load_tileA(H1 + mt * 16 * 64, 64, lane);
                v16h a1 = load_tileA(H1 + mt * 16 * 64 + 32, 64, lane);
#pragma unroll
                for (int ntile = 0; ntile < 2; ++ntile) {
                    v8f acc = wmma16(a0, wb[ntile * 2], vzero8());
                    acc = wmma16(a1, wb[ntile * 2 + 1], acc);
                    store_tile_T_bn(T2t + ntile * 16 * 96 + mt * 16, 96, acc,
                                    nullptr, 0, false, 16, lane);
                }
            }
        }

        // GEMM4: H2 = relu(Â @ T2 + b2)^T -> [feat(32)][node(96)] conv input
        for (int mt = 0; mt < 6; ++mt) {
            v16h a0 = load_tileA(sAhat + mt * 16 * 96, 96, lane);
            v16h a1 = load_tileA(sAhat + mt * 16 * 96 + 32, 96, lane);
            v16h a2 = load_tileA(sAhat + mt * 16 * 96 + 64, 96, lane);
#pragma unroll
            for (int ntile = 0; ntile < 2; ++ntile) {
                v8f acc = wmma16(a0, load_tileB(T2t + ntile * 16 * 96, 96, lane), vzero8());
                acc = wmma16(a1, load_tileB(T2t + ntile * 16 * 96 + 32, 96, lane), acc);
                acc = wmma16(a2, load_tileB(T2t + ntile * 16 * 96 + 64, 96, lane), acc);
                store_tile_T_bn(H2 + ntile * 16 * 96 + mt * 16, 96, acc,
                                gcn2_b, ntile * 16, true, 16, lane);
            }
        }

        // conv stack
        conv_layer<32, 96, 64, 48, 7, 2, 3>(H2, U + U_IC1, U + U_C1O, wsC1, conv1_b, lane);
        conv_layer<64, 48, 128, 24, 5, 2, 2>(U + U_C1O, U + U_IC2, U + U_C2O, wsC2, conv2_b, lane);
        conv_layer<128, 24, 256, 12, 3, 2, 1>(U + U_C2O, U + U_IC3, sA4 + arow * 3072,
                                              wsC3, conv3_b, lane);
    }
    __syncthreads();

    // ---- phase 2: fc1 (M=16 batches, K=3072, N=128), 1 N-tile per wave ----
    {
        int n0 = wv * 16;
        v8f acc = vzero8();
        for (int kt = 0; kt < 96; ++kt) {
            v16h a = load_tileA(sA4 + kt * 32, 3072, lane);
            v16h bm = load_tileB(wsF1 + n0 * 3072 + kt * 32, 3072, lane);
            acc = wmma16(a, bm, acc);
            __builtin_prefetch((const void*)(wsF1 + n0 * 3072 + kt * 32 + 512), 0, 0);
        }
        store_tile_f32(outFeat + (size_t)b0g * 128 + n0, 128, acc, fc1_b, n0, false, 16, lane);
        store_tile_f16(sR1 + n0, 128, acc, fc1_b, n0, true, lane);
    }
    __syncthreads();

    // ---- phase 3: cls1 (K=128, N=128), 1 N-tile per wave ----
    {
        int n0 = wv * 16;
        v8f acc = vzero8();
#pragma unroll
        for (int kt = 0; kt < 4; ++kt) {
            v16h a = load_tileA(sR1 + kt * 32, 128, lane);
            v16h bm = load_tileB(wsL1 + n0 * 128 + kt * 32, 128, lane);
            acc = wmma16(a, bm, acc);
        }
        store_tile_f16(sR2 + n0, 128, acc, cls1_b, n0, true, lane);
    }
    __syncthreads();

    // ---- phase 4: cls2 (K=128, N=210 padded 224 = 14 tiles), 2 tiles per wave ----
    for (int q = 0; q < 2; ++q) {
        int ntile = wv * 2 + q;
        if (ntile >= 14) break;
        int n0 = ntile * 16;
        v8f acc = vzero8();
#pragma unroll
        for (int kt = 0; kt < 4; ++kt) {
            v16h a = load_tileA(sR2 + kt * 32, 128, lane);
            v16h bm = load_tileB(wsL2 + n0 * 128 + kt * 32, 128, lane);
            acc = wmma16(a, bm, acc);
        }
        int vc = 210 - n0;
        if (vc > 16) vc = 16;
        store_tile_f32(outCls + (size_t)b0g * 210 + n0, 210, acc, cls2_b, n0, false, vc, lane);
    }
}

// ---------------- host launcher ----------------
extern "C" void kernel_launch(void* const* d_in, const int* in_sizes, int n_in,
                              void* d_out, int out_size, void* d_ws, size_t ws_size,
                              hipStream_t stream) {
    (void)in_sizes; (void)n_in; (void)out_size; (void)ws_size;
    const float* src     = (const float*)d_in[0];
    const int*   edge    = (const int*)d_in[1];
    const float* gcn1_w  = (const float*)d_in[2];
    const float* gcn1_b  = (const float*)d_in[3];
    const float* gcn2_w  = (const float*)d_in[4];
    const float* gcn2_b  = (const float*)d_in[5];
    const float* conv1_w = (const float*)d_in[6];
    const float* conv1_b = (const float*)d_in[7];
    const float* conv2_w = (const float*)d_in[8];
    const float* conv2_b = (const float*)d_in[9];
    const float* conv3_w = (const float*)d_in[10];
    const float* conv3_b = (const float*)d_in[11];
    const float* fc1_w   = (const float*)d_in[12];
    const float* fc1_b   = (const float*)d_in[13];
    const float* cls1_w  = (const float*)d_in[14];
    const float* cls1_b  = (const float*)d_in[15];
    const float* cls2_w  = (const float*)d_in[16];
    const float* cls2_b  = (const float*)d_in[17];

    half_t* wsh = (half_t*)d_ws;
    float* outCls = (float*)d_out;
    float* outFeat = outCls + (size_t)NB * 210;

    k_prep_adj<<<1, 128, 0, stream>>>(edge, wsh + OFF_AHAT);
    k_prep_weights<<<256, 256, 0, stream>>>(gcn1_w, gcn2_w, conv1_w, conv2_w, conv3_w,
                                            fc1_w, cls1_w, cls2_w, wsh);

    (void)hipFuncSetAttribute((const void*)k_fused,
                              hipFuncAttributeMaxDynamicSharedMemorySize, (int)SMEM_BYTES);
    k_fused<<<NB / 16, 256, SMEM_BYTES, stream>>>(src, gcn1_b, gcn2_b, conv1_b, conv2_b,
                                                  conv3_b, fc1_b, cls1_b, cls2_b, wsh,
                                                  outCls, outFeat);
}
